// Similarity_Loss_Sigmoid_Cyclic_23794118820235
// MI455X (gfx1250) — compile-verified, matrix operand order checked by reading, not run on hardware
//
#include <hip/hip_runtime.h>
#include <hip/hip_bf16.h>

// ---------------------------------------------------------------------------
// SigLIP sigmoid loss:  out = sum_ij softplus(label_ij * (bias - 10*sim_ij)) / B
//   sim = normalize(img) @ normalize(txt)^T   (B=8192, D=1024)
// fp32 normalize -> bf16, v_wmma_f32_16x16x32_bf16 GEMM with:
//   - B-tile staged to LDS via gfx1250 async-to-LDS (ASYNCcnt), double-buffered
//   - sched_barrier-pinned bulk B-fragment LDS loads (staggered dscnt waits)
//   - A fragment software-pipelined one K-step ahead (L2 latency hidden)
//   - fused fast-softplus epilogue, deterministic two-stage reduction
// ---------------------------------------------------------------------------

#define B_DIM 8192
#define D_DIM 1024
#define TEMP  10.0f

typedef __attribute__((ext_vector_type(16))) __bf16 v16bf;
typedef __attribute__((ext_vector_type(8)))  float  v8f;

struct U128x2 { uint4 lo; uint4 hi; };     // 32 bytes == v16bf
struct BF16x4 { __bf16 h[4]; };            // 8 bytes  == uint2

// gfx1250 async global->LDS copy of 16 bytes per lane (tracked by ASYNCcnt).
__device__ __forceinline__ void async_copy_b128(unsigned lds_off, const void* gaddr) {
    asm volatile("global_load_async_to_lds_b128 %0, %1, off"
                 :: "v"(lds_off), "v"(gaddr) : "memory");
}
__device__ __forceinline__ void wait_async_le2() {
    asm volatile("s_wait_asynccnt 0x2" ::: "memory");
}
__device__ __forceinline__ void wait_async_le0() {
    asm volatile("s_wait_asynccnt 0x0" ::: "memory");
}
__device__ __forceinline__ void sched_fence() {
#if __has_builtin(__builtin_amdgcn_sched_barrier)
    __builtin_amdgcn_sched_barrier(0);
#endif
}

// ---------------------------------------------------------------------------
// Kernel 1: row L2-normalize (fp32) and convert to bf16.
// ---------------------------------------------------------------------------
__global__ __launch_bounds__(256)
void normalize_bf16_kernel(const float* __restrict__ txt,
                           const float* __restrict__ img,
                           __bf16* __restrict__ imgN,
                           __bf16* __restrict__ txtN) {
    const int rowId = blockIdx.x;
    const bool isImg = rowId < B_DIM;
    const int r = isImg ? rowId : (rowId - B_DIM);
    const float*  src = isImg ? img  : txt;
    __bf16*       dst = isImg ? imgN : txtN;

    const float4* p = (const float4*)(src + (size_t)r * D_DIM);
    float4 x = p[threadIdx.x];
    float s = x.x * x.x + x.y * x.y + x.z * x.z + x.w * x.w;

    #pragma unroll
    for (int o = 16; o > 0; o >>= 1) s += __shfl_xor(s, o, 32);

    __shared__ float red[8];
    const int lane = threadIdx.x & 31;
    const int wave = threadIdx.x >> 5;
    if (lane == 0) red[wave] = s;
    __syncthreads();
    if (wave == 0) {
        float t = (lane < 8) ? red[lane] : 0.0f;
        #pragma unroll
        for (int o = 4; o > 0; o >>= 1) t += __shfl_xor(t, o, 32);
        if (lane == 0) red[0] = t;
    }
    __syncthreads();

    const float inv = 1.0f / fmaxf(sqrtf(red[0]), 1e-12f);

    BF16x4 pk;
    pk.h[0] = (__bf16)(x.x * inv);
    pk.h[1] = (__bf16)(x.y * inv);
    pk.h[2] = (__bf16)(x.z * inv);
    pk.h[3] = (__bf16)(x.w * inv);
    *(uint2*)(dst + (size_t)r * D_DIM + (size_t)threadIdx.x * 4) =
        __builtin_bit_cast(uint2, pk);
}

// ---------------------------------------------------------------------------
// Kernel 2: 128x128 tile per workgroup, 8 waves, wave w owns rows
// [tileM+16w, +16) x 128 cols -> 8 v8f accumulators.
//
// B-tile (128 cols x 32 K of bf16 = 8 KB) staged to LDS, double-buffered:
//   LDS layout: column c's 32 K-values contiguous at byte offset c*64.
//   Fragment (col group t): lane L reads 16B at col*64 + (L>>4)*16 and +32.
// A-fragment: direct global, lane L: row = tileM+16w+(L&15),
//   chunks at k0 = k + 8*(L>>4) and k0+16 (two b128 loads), prefetched
//   one K-step ahead.
// C/D layout: VGPR r, lane L -> M = r + 8*(L>>4), N = L & 15.
// ---------------------------------------------------------------------------
__global__ __launch_bounds__(256)
void gemm_loss_kernel(const __bf16* __restrict__ imgN,
                      const __bf16* __restrict__ txtN,
                      const float* __restrict__ bias,
                      float* __restrict__ partials) {
    const int tid  = threadIdx.x;
    const int lane = tid & 31;
    const int wave = tid >> 5;
    const int half = lane >> 4;      // 0 or 1
    const int l15  = lane & 15;

    const int tileM = blockIdx.y * 128;
    const int tileN = blockIdx.x * 128;

    // 2 x 8KB double buffer for the B tile, as uint4 chunks (512 per buffer).
    __shared__ uint4 bsm[2][512];

    // Staging addressing: this thread copies chunks q0=tid*2 and q0+1.
    // chunk q -> col = q>>2, part = q&3; 16 bytes from txtN[tileN+col][k + part*8].
    const int q0   = tid * 2;
    const int col0 = q0 >> 2, part0 = q0 & 3;
    const int col1 = (q0 + 1) >> 2, part1 = (q0 + 1) & 3;
    const char* tbase = (const char*)txtN;
    const unsigned trow0 = (unsigned)(tileN + col0) * (D_DIM * 2) + part0 * 16;
    const unsigned trow1 = (unsigned)(tileN + col1) * (D_DIM * 2) + part1 * 16;
    const unsigned lds0 = (unsigned)(size_t)&bsm[0][q0];
    const unsigned lds1 = (unsigned)(size_t)&bsm[1][q0];

    // A fragment addressing (32-bit byte offsets off uniform base).
    const char* abase = (const char*)imgN;
    const unsigned arow = (unsigned)(tileM + wave * 16 + l15) * (D_DIM * 2)
                        + (unsigned)half * 16;   // +k*2 added per step

    v8f c[8];
    #pragma unroll
    for (int t = 0; t < 8; ++t) c[t] = (v8f)0.0f;

    // Prologue: stage k=0 B-tile into buffer 0; prefetch k=0 A fragment.
    async_copy_b128(lds0,      tbase + trow0);
    async_copy_b128(lds0 + 16, tbase + trow1);
    U128x2 araw;
    araw.lo = *(const uint4*)(abase + arow);
    araw.hi = *(const uint4*)(abase + arow + 32);

    for (int k = 0; k < D_DIM; k += 32) {
        const int buf = (k >> 5) & 1;
        const unsigned kb = (unsigned)k * 2;   // byte offset of k within a row
        const bool more = (k + 32 < D_DIM);

        if (more) {             // stage next B-tile into the other buffer
            const unsigned nb = kb + 64;
            const unsigned ldsN = buf ? lds0 : lds1;
            async_copy_b128(ldsN,      tbase + trow0 + nb);
            async_copy_b128(ldsN + 16, tbase + trow1 + nb);
            wait_async_le2();   // current buffer's 2 copies complete
        } else {
            wait_async_le0();
        }
        __syncthreads();        // all waves' staging of current buffer visible

        // Prefetch next A fragment (consumed next iteration -> latency hidden).
        const unsigned nkb = more ? (kb + 64) : kb;
        U128x2 araw_next;
        araw_next.lo = *(const uint4*)(abase + arow + nkb);
        araw_next.hi = *(const uint4*)(abase + arow + nkb + 32);

        // Bulk-load all 8 B fragments (16 ds_load_b128), then fence the
        // scheduler so WMMAs issue against staggered dscnt waits.
        U128x2 braw[8];
        #pragma unroll
        for (int t = 0; t < 8; ++t) {
            const int ci = (t * 16 + l15) * 4 + half;   // col*4 + half
            braw[t].lo = bsm[buf][ci];
            braw[t].hi = bsm[buf][ci + 2];
        }
        sched_fence();

        const v16bf a = __builtin_bit_cast(v16bf, araw);
        #pragma unroll
        for (int t = 0; t < 8; ++t) {
            const v16bf b = __builtin_bit_cast(v16bf, braw[t]);
            c[t] = __builtin_amdgcn_wmma_f32_16x16x32_bf16(
                       false, a, false, b, (short)0, c[t], false, false);
        }
        araw = araw_next;
        __syncthreads();        // reads done before buffer is overwritten
    }

    // Fused epilogue: softplus(label * (bias - TEMP*sim)) via fast TRANS ops.
    const float bs = bias[0];
    float acc = 0.0f;
    #pragma unroll
    for (int t = 0; t < 8; ++t) {
        const int n = tileN + t * 16 + l15;
        #pragma unroll
        for (int r = 0; r < 8; ++r) {
            const int m = tileM + wave * 16 + half * 8 + r;
            const float logit = bs - TEMP * c[t][r];
            const float x = (m == n) ? logit : -logit;
            // stable softplus: max(x,0) + log(1 + exp(-|x|))
            acc += fmaxf(x, 0.0f) + __logf(1.0f + __expf(-fabsf(x)));
        }
    }

    #pragma unroll
    for (int o = 16; o > 0; o >>= 1) acc += __shfl_xor(acc, o, 32);
    __shared__ float red[8];
    if (lane == 0) red[wave] = acc;
    __syncthreads();
    if (tid == 0) {
        float s = 0.0f;
        #pragma unroll
        for (int i = 0; i < 8; ++i) s += red[i];
        partials[blockIdx.y * gridDim.x + blockIdx.x] = s;
    }
}

// ---------------------------------------------------------------------------
// Kernel 3: fixed-order reduction of 4096 partials, divide by B.
// ---------------------------------------------------------------------------
__global__ __launch_bounds__(256)
void final_reduce_kernel(const float* __restrict__ partials,
                         float* __restrict__ out) {
    float s = 0.0f;
    for (int i = threadIdx.x; i < 4096; i += 256) s += partials[i];
    #pragma unroll
    for (int o = 16; o > 0; o >>= 1) s += __shfl_xor(s, o, 32);
    __shared__ float red[8];
    const int lane = threadIdx.x & 31;
    const int wave = threadIdx.x >> 5;
    if (lane == 0) red[wave] = s;
    __syncthreads();
    if (threadIdx.x == 0) {
        float tot = 0.0f;
        #pragma unroll
        for (int i = 0; i < 8; ++i) tot += red[i];
        out[0] = tot / (float)B_DIM;
    }
}

// ---------------------------------------------------------------------------
extern "C" void kernel_launch(void* const* d_in, const int* in_sizes, int n_in,
                              void* d_out, int out_size, void* d_ws, size_t ws_size,
                              hipStream_t stream) {
    (void)in_sizes; (void)n_in; (void)out_size; (void)ws_size;

    const float* txt  = (const float*)d_in[0];   // text_embeddings  [B, D]
    const float* img  = (const float*)d_in[1];   // image_embeddings [B, D]
    const float* bias = (const float*)d_in[2];   // bias [1]
    float* out = (float*)d_out;

    // Workspace: imgN bf16 (16 MB) | txtN bf16 (16 MB) | partials (16 KB)
    __bf16* imgN = (__bf16*)d_ws;
    __bf16* txtN = imgN + (size_t)B_DIM * D_DIM;
    float* partials = (float*)((char*)d_ws + 2ull * B_DIM * D_DIM * sizeof(__bf16));

    normalize_bf16_kernel<<<2 * B_DIM, 256, 0, stream>>>(txt, img, imgN, txtN);

    dim3 grid(B_DIM / 128, B_DIM / 128);   // 64 x 64 tiles
    gemm_loss_kernel<<<grid, 256, 0, stream>>>(imgN, txtN, bias, partials);

    final_reduce_kernel<<<1, 256, 0, stream>>>(partials, out);
}
